// BiasedAttentionTransformer_81612968558725
// MI455X (gfx1250) — matrix operand order repaired and checked by reading, not compiled, not run on hardware
//
#include <hip/hip_runtime.h>

// ---------------- problem constants ----------------
constexpr int B_  = 64;
constexpr int L_  = 256;
constexpr int E_  = 768;
constexpr int H_  = 12;
constexpr int DH_ = 64;
constexpr int D_  = 12;
constexpr int M_  = B_ * L_;   // 16384 token rows

typedef __attribute__((ext_vector_type(16))) _Float16 v16h;
typedef __attribute__((ext_vector_type(8)))  float    v8f;

// ---------------- WMMA fragment helpers (contiguous k order: k = 16*(l/16) + s) ----------------
// A 16x32 f16 fragment: lane l holds row m=l%16; K-base = (l/16)*16.
// VGPR i (halves 2i,2i+1): K = kbase + 2i  (+1 for odd half)
__device__ __forceinline__ v16h load_a_frag(const _Float16* __restrict__ base,
                                            int row0, int ldm) {
  int lane = threadIdx.x & 31;
  int m  = lane & 15;
  int kb = (lane >> 4) << 4;
  const _Float16* p = base + (size_t)(row0 + m) * ldm;
  v16h a;
#pragma unroll
  for (int i = 0; i < 8; ++i) {
    int k = kb + 2 * i;
    a[2 * i]     = p[k];
    a[2 * i + 1] = p[k + 1];
  }
  return a;
}

// B 32x16 f16 fragment: lane l holds col n=l%16; K-base = (l/16)*16.
// VGPR i: K = kbase + 2i (and +1).
__device__ __forceinline__ v16h load_b_frag(const _Float16* __restrict__ base,
                                            int col0, int ldn) {
  int lane = threadIdx.x & 31;
  int n   = lane & 15;
  int kb2 = (lane >> 4) << 4;
  const _Float16* p = base + col0 + n;
  v16h b;
#pragma unroll
  for (int i = 0; i < 8; ++i) {
    int k = kb2 + 2 * i;
    b[2 * i]     = p[(size_t)k * ldn];
    b[2 * i + 1] = p[(size_t)(k + 1) * ldn];
  }
  return b;
}

// ---------------- small utility kernels ----------------
__global__ void cvt_f16_kernel(const float* __restrict__ src,
                               _Float16* __restrict__ dst, size_t n) {
  size_t i = (size_t)blockIdx.x * blockDim.x + threadIdx.x;
  if (i < n) dst[i] = (_Float16)src[i];
}

__global__ void embed_kernel(const int* __restrict__ tokens,
                             const float* __restrict__ embed,
                             float* __restrict__ e) {
  size_t idx = (size_t)blockIdx.x * blockDim.x + threadIdx.x;
  size_t total = (size_t)M_ * (E_ / 4);
  if (idx >= total) return;
  int row = (int)(idx / (E_ / 4));
  int c4  = (int)(idx % (E_ / 4)) * 4;
  int tok = tokens[row];
  *(float4*)(e + (size_t)row * E_ + c4) =
      *(const float4*)(embed + (size_t)tok * E_ + c4);
}

__global__ void dist_kernel(const float* __restrict__ r,
                            float* __restrict__ dmat) {
  size_t idx = (size_t)blockIdx.x * blockDim.x + threadIdx.x;
  size_t total = (size_t)B_ * L_ * L_;
  if (idx >= total) return;
  int j = (int)(idx % L_);
  int i = (int)((idx / L_) % L_);
  int b = (int)(idx / ((size_t)L_ * L_));
  const float* ri = r + ((size_t)b * L_ + i) * 3;
  const float* rj = r + ((size_t)b * L_ + j) * 3;
  float dx = ri[0] - rj[0], dy = ri[1] - rj[1], dz = ri[2] - rj[2];
  dmat[idx] = sqrtf(dx * dx + dy * dy + dz * dz + 1e-12f);
}

// LayerNorm: one wave per row (768 cols -> 24 per lane), shuffle reduction.
template <bool WRITE_F16>
__global__ __launch_bounds__(256) void ln_kernel(
    const float* __restrict__ X, const float* __restrict__ g,
    const float* __restrict__ bb, float* __restrict__ Yf32,
    _Float16* __restrict__ Yf16, int N) {
  int wave = threadIdx.x >> 5, lane = threadIdx.x & 31;
  size_t row = (size_t)blockIdx.x * 8 + wave;
  const float* x = X + row * N;
  float s = 0.f, ss = 0.f;
  for (int c = lane; c < N; c += 32) {
    float v = x[c];
    s += v;
    ss += v * v;
  }
#pragma unroll
  for (int off = 1; off < 32; off <<= 1) {
    s  += __shfl_xor(s, off);
    ss += __shfl_xor(ss, off);
  }
  float invN = 1.0f / (float)N;
  float mean = s * invN;
  float var  = ss * invN - mean * mean;
  float inv  = rsqrtf(fmaxf(var, 0.f) + 1e-5f);
  for (int c = lane; c < N; c += 32) {
    float y = (x[c] - mean) * inv * g[c] + bb[c];
    Yf32[row * N + c] = y;
    if (WRITE_F16) Yf16[row * N + c] = (_Float16)y;
  }
}

// ---------------- tiled WMMA GEMM ----------------
// C[M,N] = A[M,K](f16) @ Bm[K,N](f16) + bias, compile-time epilogue.
// Block: 128 threads (4 waves), 64x128 tile; wave -> 32x64 (eight 16x16 accs).
template <bool RELU, bool HAS_RESID, bool HAS_MASK, bool OUT_F32, bool OUT_F16>
__global__ __launch_bounds__(128) void gemm_f16_kernel(
    const _Float16* __restrict__ A, const _Float16* __restrict__ Bm,
    const float* __restrict__ bias, const float* __restrict__ resid,
    const unsigned char* __restrict__ rowmask, float* __restrict__ Cf32,
    _Float16* __restrict__ Cf16, int M, int N, int K) {
  __shared__ __align__(16) _Float16 sA[64][40];   // padded: conflict-free
  __shared__ __align__(16) _Float16 sB[32][136];
  int m0 = blockIdx.y * 64, n0 = blockIdx.x * 128;
  int tid = threadIdx.x;
  int wave = tid >> 5, lane = tid & 31;
  int wm = wave >> 1, wn = wave & 1;   // wave tile: rows wm*32, cols wn*64
  v8f acc[2][4] = {};

  for (int k0 = 0; k0 < K; k0 += 32) {
    __syncthreads();
#pragma unroll
    for (int it = 0; it < 2; ++it) {  // A tile: 64x32 halves, 8-half chunks
      int chunk = tid + it * 128;
      int row = chunk >> 2, cc = (chunk & 3) << 3;
      *(uint4*)(&sA[row][cc]) =
          *(const uint4*)(A + (size_t)(m0 + row) * K + k0 + cc);
    }
#pragma unroll
    for (int it = 0; it < 4; ++it) {  // B tile: 32x128 halves
      int chunk = tid + it * 128;
      int row = chunk >> 4, cc = (chunk & 15) << 3;
      *(uint4*)(&sB[row][cc]) =
          *(const uint4*)(Bm + (size_t)(k0 + row) * N + n0 + cc);
    }
    if (k0 + 32 < K) {  // prefetch next k-tile (global_prefetch_b8)
      int rowa = tid >> 2, cca = (tid & 3) << 3;
      __builtin_prefetch(A + (size_t)(m0 + rowa) * K + k0 + 32 + cca, 0, 3);
      int rowb = tid >> 4, ccb = (tid & 15) << 3;
      __builtin_prefetch(Bm + (size_t)(k0 + 32 + rowb) * N + n0 + ccb, 0, 3);
    }
    __syncthreads();
    v16h af[2], bf[4];
    af[0] = load_a_frag(&sA[0][0], wm * 32, 40);
    af[1] = load_a_frag(&sA[0][0], wm * 32 + 16, 40);
#pragma unroll
    for (int ni = 0; ni < 4; ++ni)
      bf[ni] = load_b_frag(&sB[0][0], wn * 64 + ni * 16, 136);
#pragma unroll
    for (int mi = 0; mi < 2; ++mi)
#pragma unroll
      for (int ni = 0; ni < 4; ++ni)
        acc[mi][ni] = __builtin_amdgcn_wmma_f32_16x16x32_f16(
            false, af[mi], false, bf[ni], (short)0, acc[mi][ni], false, false);
  }

  int rb = (lane >> 4) << 3;
  int cn = lane & 15;
  float bcol[4];
#pragma unroll
  for (int ni = 0; ni < 4; ++ni) bcol[ni] = bias[n0 + wn * 64 + ni * 16 + cn];
#pragma unroll
  for (int mi = 0; mi < 2; ++mi)
#pragma unroll
    for (int rr = 0; rr < 8; ++rr) {
      int row = m0 + wm * 32 + mi * 16 + rb + rr;
      bool mz = false;
      if (HAS_MASK) mz = rowmask[row] != 0;
      const float* rrow = HAS_RESID ? (resid + (size_t)row * N) : nullptr;
#pragma unroll
      for (int ni = 0; ni < 4; ++ni) {
        int col = n0 + wn * 64 + ni * 16 + cn;
        float v = acc[mi][ni][rr] + bcol[ni];
        if (RELU) v = fmaxf(v, 0.0f);
        if (HAS_MASK) v = mz ? 0.0f : v;   // pad-mask before residual
        if (HAS_RESID) v += rrow[col];
        size_t o = (size_t)row * N + col;
        if (OUT_F32) Cf32[o] = v;
        if (OUT_F16) Cf16[o] = (_Float16)v;
      }
    }
}

// ---------------- fused flash attention (one block per (b,h)) ----------------
// 8 waves x 32 query rows; K/V blocks staged cooperatively in LDS (coalesced
// uint4 loads); S = Q K^T via WMMA, Gaussian bias + mask, online softmax,
// P restaged via LDS into A-fragment form, O += P V via WMMA.
__global__ __launch_bounds__(256) void attn_kernel(
    const _Float16* __restrict__ qkv,   // [M, 3E] f16
    const float* __restrict__ dmat,     // [B, L, L]
    const unsigned char* __restrict__ pad,  // [B, L]
    const float* __restrict__ muA, const float* __restrict__ gaA,
    const float* __restrict__ wA, int layer,
    _Float16* __restrict__ outO)        // [M, E] f16
{
  __shared__ __align__(16) _Float16 sP[8][32][40];
  __shared__ __align__(16) _Float16 sK[32][72];  // key block, row-major (j, dh)
  __shared__ __align__(16) _Float16 sV[32][72];  // value block, row-major (j, dh)
  const int bh = blockIdx.x;
  const int b = bh / H_, h = bh - b * H_;
  const int tid = threadIdx.x;
  const int wave = tid >> 5, lane = tid & 31;
  const int i0 = wave * 32;
  const float muv = muA[layer * H_ + h];
  const float gav = gaA[layer * H_ + h];
  const float wv  = wA[layer * H_ + h];
  const float scale = 0.125f;  // 1/sqrt(64)
  const size_t rowbase = (size_t)b * L_;
  const int m   = lane & 15;
  const int kb  = (lane >> 4) << 3;
  const int kb2 = (lane >> 4) << 4;
  const int rb  = (lane >> 4) << 3;
  const int cn  = lane & 15;
  // cooperative K/V tile coords: 32 rows x 64 halves = 256 8-half chunks
  const int ldrow = tid >> 3;
  const int ldcc  = (tid & 7) << 3;

  // Q fragments (2 row-tiles x 2 k-steps) gathered from global (once).
  v16h aq[2][2];
#pragma unroll
  for (int mi = 0; mi < 2; ++mi)
#pragma unroll
    for (int kk = 0; kk < 2; ++kk) {
      const _Float16* qp = qkv + (rowbase + i0 + mi * 16 + m) * (size_t)(3 * E_) +
                           h * DH_ + kk * 32;
#pragma unroll
      for (int i = 0; i < 8; ++i) {
        int k = kb2 + 2 * i;
        aq[mi][kk][2 * i]     = qp[k];
        aq[mi][kk][2 * i + 1] = qp[k + 1];
      }
    }

  // row pad flags for this lane's C-layout rows (hoisted out of j-loop)
  bool padI[2][8];
#pragma unroll
  for (int mi = 0; mi < 2; ++mi)
#pragma unroll
    for (int rr = 0; rr < 8; ++rr)
      padI[mi][rr] = pad[rowbase + i0 + mi * 16 + rb + rr] != 0;

  float mrow[2][8], lrow[2][8];
  v8f oacc[2][4] = {};
#pragma unroll
  for (int mi = 0; mi < 2; ++mi)
#pragma unroll
    for (int rr = 0; rr < 8; ++rr) {
      mrow[mi][rr] = -1e30f;
      lrow[mi][rr] = 0.0f;
    }

  for (int j0 = 0; j0 < L_; j0 += 32) {
    // ---- stage K/V blocks into LDS (coalesced) ----
    __syncthreads();  // previous iteration done reading sK/sV
    {
      const _Float16* kg = qkv + (rowbase + j0 + ldrow) * (size_t)(3 * E_) +
                           E_ + h * DH_ + ldcc;
      *(uint4*)(&sK[ldrow][ldcc]) = *(const uint4*)kg;
      const _Float16* vg = qkv + (rowbase + j0 + ldrow) * (size_t)(3 * E_) +
                           2 * E_ + h * DH_ + ldcc;
      *(uint4*)(&sV[ldrow][ldcc]) = *(const uint4*)vg;
    }
    __syncthreads();

    // ---- S = Q K^T  (K B-fragment = contiguous dh run in sK row) ----
    v8f sacc[2][2] = {};
#pragma unroll
    for (int kk = 0; kk < 2; ++kk)
#pragma unroll
      for (int ni = 0; ni < 2; ++ni) {
        v16h bk;
        const _Float16* kp = &sK[ni * 16 + cn][kk * 32 + kb2];
#pragma unroll
        for (int i = 0; i < 16; ++i) bk[i] = kp[i];
#pragma unroll
        for (int mi = 0; mi < 2; ++mi)
          sacc[mi][ni] = __builtin_amdgcn_wmma_f32_16x16x32_f16(
              false, aq[mi][kk], false, bk, (short)0, sacc[mi][ni], false, false);
      }

    // ---- scale + Gaussian bias + mask ----
    bool padJ[2];
#pragma unroll
    for (int ni = 0; ni < 2; ++ni)
      padJ[ni] = pad[rowbase + j0 + ni * 16 + cn] != 0;
#pragma unroll
    for (int mi = 0; mi < 2; ++mi)
#pragma unroll
      for (int ni = 0; ni < 2; ++ni)
#pragma unroll
        for (int rr = 0; rr < 8; ++rr) {
          int ii = i0 + mi * 16 + rb + rr;
          int jj = j0 + ni * 16 + cn;
          float s = sacc[mi][ni][rr] * scale;
          float dd = dmat[((size_t)b * L_ + ii) * L_ + jj];
          float t = dd - muv;
          s += wv * __expf(-gav * t * t);
          bool mk = (ii == jj) || padI[mi][rr] || padJ[ni];
          sacc[mi][ni][rr] = mk ? -1.0e9f : s;
        }

    // ---- online softmax (row reduction within 16-lane halves) ----
#pragma unroll
    for (int mi = 0; mi < 2; ++mi)
#pragma unroll
      for (int rr = 0; rr < 8; ++rr) {
        float mx = fmaxf(sacc[mi][0][rr], sacc[mi][1][rr]);
#pragma unroll
        for (int off = 1; off < 16; off <<= 1) mx = fmaxf(mx, __shfl_xor(mx, off));
        float mold = mrow[mi][rr];
        float mnew = fmaxf(mold, mx);
        float alpha = __expf(mold - mnew);
        float e0v = __expf(sacc[mi][0][rr] - mnew);
        float e1v = __expf(sacc[mi][1][rr] - mnew);
        sacc[mi][0][rr] = e0v;
        sacc[mi][1][rr] = e1v;
        float rs = e0v + e1v;
#pragma unroll
        for (int off = 1; off < 16; off <<= 1) rs += __shfl_xor(rs, off);
        mrow[mi][rr] = mnew;
        lrow[mi][rr] = lrow[mi][rr] * alpha + rs;
#pragma unroll
        for (int nj = 0; nj < 4; ++nj) oacc[mi][nj][rr] *= alpha;
      }

    // ---- restage P (C-layout -> A-fragment layout) via LDS ----
#pragma unroll
    for (int mi = 0; mi < 2; ++mi)
#pragma unroll
      for (int ni = 0; ni < 2; ++ni)
#pragma unroll
        for (int rr = 0; rr < 8; ++rr)
          sP[wave][mi * 16 + rb + rr][ni * 16 + cn] = (_Float16)sacc[mi][ni][rr];
    __syncthreads();
    v16h pf[2];
#pragma unroll
    for (int mi = 0; mi < 2; ++mi) {
      const _Float16* pp = &sP[wave][mi * 16 + m][0];
#pragma unroll
      for (int i = 0; i < 8; ++i) {
        int k = kb2 + 2 * i;
        pf[mi][2 * i]     = pp[k];
        pf[mi][2 * i + 1] = pp[k + 1];
      }
    }

    // ---- O += P @ V  (V B-fragment from sV, padded stride: conflict-free) ----
#pragma unroll
    for (int nj = 0; nj < 4; ++nj) {
      v16h vf;
      const _Float16* vp = &sV[kb2][nj * 16 + cn];
#pragma unroll
      for (int i = 0; i < 16; ++i) vf[i] = vp[(size_t)i * 72];
#pragma unroll
      for (int mi = 0; mi < 2; ++mi)
        oacc[mi][nj] = __builtin_amdgcn_wmma_f32_16x16x32_f16(
            false, pf[mi], false, vf, (short)0, oacc[mi][nj], false, false);
    }
  }

  // ---- normalize + store (f16, feeds Wo GEMM) ----
#pragma unroll
  for (int mi = 0; mi < 2; ++mi)
#pragma unroll
    for (int nj = 0; nj < 4; ++nj)
#pragma unroll
      for (int rr = 0; rr < 8; ++rr) {
        int ii = i0 + mi * 16 + rb + rr;
        int col = h * DH_ + nj * 16 + cn;
        float v = oacc[mi][nj][rr] / lrow[mi][rr];
        outO[(rowbase + ii) * (size_t)E_ + col] = (_Float16)v;
      }
}

// ---------------- final head ----------------
__global__ __launch_bounds__(256) void colmean_kernel(const float* __restrict__ X,
                                                      float* __restrict__ Mn) {
  int b = blockIdx.x;
  for (int c = threadIdx.x; c < E_; c += 256) {
    float s = 0.f;
    for (int l = 0; l < L_; ++l) s += X[((size_t)b * L_ + l) * E_ + c];
    Mn[(size_t)b * E_ + c] = s * (1.0f / (float)L_);
  }
}

__global__ __launch_bounds__(256) void head_kernel(const float* __restrict__ Mn,
                                                   const float* __restrict__ Wout,
                                                   const float* __restrict__ bout,
                                                   float* __restrict__ out) {
  __shared__ float red[256];
  int b = blockIdx.x;
  float acc = 0.f;
  for (int c = threadIdx.x; c < E_; c += 256)
    acc += Mn[(size_t)b * E_ + c] * Wout[c];  // OUT == 1
  red[threadIdx.x] = acc;
  __syncthreads();
  for (int s = 128; s > 0; s >>= 1) {
    if (threadIdx.x < s) red[threadIdx.x] += red[threadIdx.x + s];
    __syncthreads();
  }
  if (threadIdx.x == 0) out[b] = red[0] + bout[0];
}

// ---------------- host orchestration ----------------
extern "C" void kernel_launch(void* const* d_in, const int* in_sizes, int n_in,
                              void* d_out, int out_size, void* d_ws,
                              size_t ws_size, hipStream_t stream) {
  (void)in_sizes; (void)n_in; (void)out_size; (void)ws_size;
  const int*           tokens  = (const int*)d_in[0];
  const unsigned char* padding = (const unsigned char*)d_in[1];
  const float* r     = (const float*)d_in[2];
  const float* embed = (const float*)d_in[3];
  const float* mu    = (const float*)d_in[4];
  const float* gamma = (const float*)d_in[5];
  const float* w     = (const float*)d_in[6];
  const float* Wqkv  = (const float*)d_in[7];
  const float* bqkv  = (const float*)d_in[8];
  const float* Wo    = (const float*)d_in[9];
  const float* bo    = (const float*)d_in[10];
  const float* ln1_g = (const float*)d_in[11];
  const float* ln1_b = (const float*)d_in[12];
  const float* W1    = (const float*)d_in[13];
  const float* b1    = (const float*)d_in[14];
  const float* W2    = (const float*)d_in[15];
  const float* b2    = (const float*)d_in[16];
  const float* ln2_g = (const float*)d_in[17];
  const float* ln2_b = (const float*)d_in[18];
  const float* lnf_g = (const float*)d_in[19];
  const float* lnf_b = (const float*)d_in[20];
  const float* Wout  = (const float*)d_in[21];
  const float* bout  = (const float*)d_in[22];
  float* out = (float*)d_out;

  char* p = (char*)d_ws;
  auto alloc = [&](size_t bytes) -> char* {
    char* q = p;
    p += (bytes + 255) & ~(size_t)255;
    return q;
  };
  float*    dmat    = (float*)alloc((size_t)B_ * L_ * L_ * 4);
  float*    e_buf   = (float*)alloc((size_t)M_ * E_ * 4);
  float*    e0f32   = (float*)alloc((size_t)M_ * E_ * 4);
  float*    e1r     = (float*)alloc((size_t)M_ * E_ * 4);
  float*    e2f32   = (float*)alloc((size_t)M_ * E_ * 4);
  float*    meanbuf = (float*)alloc((size_t)B_ * E_ * 4);
  _Float16* e0f16   = (_Float16*)alloc((size_t)M_ * E_ * 2);
  _Float16* e2f16   = (_Float16*)alloc((size_t)M_ * E_ * 2);
  _Float16* of16    = (_Float16*)alloc((size_t)M_ * E_ * 2);
  _Float16* qkvf16  = (_Float16*)alloc((size_t)M_ * 3 * E_ * 2);
  _Float16* hf16    = (_Float16*)alloc((size_t)M_ * 4 * E_ * 2);
  _Float16* Wqkv_h  = (_Float16*)alloc((size_t)D_ * E_ * 3 * E_ * 2);
  _Float16* Wo_h    = (_Float16*)alloc((size_t)D_ * E_ * E_ * 2);
  _Float16* W1_h    = (_Float16*)alloc((size_t)D_ * E_ * 4 * E_ * 2);
  _Float16* W2_h    = (_Float16*)alloc((size_t)D_ * 4 * E_ * E_ * 2);

  auto cvt = [&](const float* s, _Float16* d, size_t n) {
    int blocks = (int)((n + 255) / 256);
    cvt_f16_kernel<<<blocks, 256, 0, stream>>>(s, d, n);
  };
  cvt(Wqkv, Wqkv_h, (size_t)D_ * E_ * 3 * E_);
  cvt(Wo,   Wo_h,   (size_t)D_ * E_ * E_);
  cvt(W1,   W1_h,   (size_t)D_ * E_ * 4 * E_);
  cvt(W2,   W2_h,   (size_t)D_ * 4 * E_ * E_);

  embed_kernel<<<((size_t)M_ * (E_ / 4) + 255) / 256, 256, 0, stream>>>(
      tokens, embed, e_buf);
  dist_kernel<<<((size_t)B_ * L_ * L_ + 255) / 256, 256, 0, stream>>>(r, dmat);

  dim3 gThr(128);
  for (int i = 0; i < D_; ++i) {
    // e0 = LN(e)  (f32 + f16 copies)
    ln_kernel<true><<<M_ / 8, 256, 0, stream>>>(
        e_buf, ln1_g + (size_t)i * E_, ln1_b + (size_t)i * E_, e0f32, e0f16, E_);
    // qkv = e0 @ Wqkv + bqkv   (f16 out)
    gemm_f16_kernel<false, false, false, false, true>
        <<<dim3(3 * E_ / 128, M_ / 64), gThr, 0, stream>>>(
            e0f16, Wqkv_h + (size_t)i * E_ * 3 * E_, bqkv + (size_t)i * 3 * E_,
            nullptr, nullptr, nullptr, qkvf16, M_, 3 * E_, E_);
    // biased masked attention -> o (f16)
    attn_kernel<<<B_ * H_, 256, 0, stream>>>(qkvf16, dmat, padding, mu, gamma, w,
                                             i, of16);
    // e1r = o @ Wo + bo + e0   (f32)
    gemm_f16_kernel<false, true, false, true, false>
        <<<dim3(E_ / 128, M_ / 64), gThr, 0, stream>>>(
            of16, Wo_h + (size_t)i * E_ * E_, bo + (size_t)i * E_, e0f32,
            nullptr, e1r, nullptr, M_, E_, E_);
    // e2 = LN(e1r)  (f32 + f16)
    ln_kernel<true><<<M_ / 8, 256, 0, stream>>>(
        e1r, ln2_g + (size_t)i * E_, ln2_b + (size_t)i * E_, e2f32, e2f16, E_);
    // h = relu(e2 @ W1 + b1)   (f16)
    gemm_f16_kernel<true, false, false, false, true>
        <<<dim3(4 * E_ / 128, M_ / 64), gThr, 0, stream>>>(
            e2f16, W1_h + (size_t)i * E_ * 4 * E_, b1 + (size_t)i * 4 * E_,
            nullptr, nullptr, nullptr, hf16, M_, 4 * E_, E_);
    // e = mask(h @ W2 + b2) + e2   (f32, overwrites e)
    gemm_f16_kernel<false, true, true, true, false>
        <<<dim3(E_ / 128, M_ / 64), gThr, 0, stream>>>(
            hf16, W2_h + (size_t)i * 4 * E_ * E_, b2 + (size_t)i * E_, e2f32,
            padding, e_buf, nullptr, M_, E_, 4 * E_);
  }

  // final LN -> mean over L -> head
  ln_kernel<false><<<M_ / 8, 256, 0, stream>>>(e_buf, lnf_g, lnf_b, e0f32,
                                               nullptr, E_);
  colmean_kernel<<<B_, 256, 0, stream>>>(e0f32, meanbuf);
  head_kernel<<<B_, 256, 0, stream>>>(meanbuf, Wout, bout, out);
}
